// DeepSet_16836271801101
// MI455X (gfx1250) — matrix operand order repaired and checked by reading, not compiled, not run on hardware
//
#include <hip/hip_runtime.h>

// ---------------------------------------------------------------------------
// DeepSet on MI455X (gfx1250, wave32, WMMA 16x16x32 f16 with f32 accumulate)
//   h   = relu(x@W1+b1)@W2+b2          [N,128]   (fused, never hits HBM)
//   s   = segment_sum(h, batch, B)     [B,128]   (run-length + atomics)
//   out = relu(s@W3+b3)@W4+b4          [B,64]
// Round 2: software-pipelined GEMM inner loop — all A fragments preloaded
// from LDS, B fragments double-buffered one k-step ahead so v_wmma never
// waits on an L0 load.
// ---------------------------------------------------------------------------

typedef _Float16 v16h __attribute__((ext_vector_type(16)));
typedef _Float16 v8h  __attribute__((ext_vector_type(8)));
typedef _Float16 v4h  __attribute__((ext_vector_type(4)));
typedef float    v8f  __attribute__((ext_vector_type(8)));

#define DDIM    128          // D_IN == D_H == D_RHO == 128
#define DOUT    64
#define PITCH   136          // LDS row pitch in halves (bank spread)
#define KTILES  4            // 128 / 32
#define ROWS    128          // rows per workgroup
#define THREADS 256          // 8 waves of 32

// --------------------------- weight packing --------------------------------
// B-operand fragment layout (ISA 7.12.2, 16-bit B 32x16, 8 VGPRs = v16h):
//   lane 0-15 : N = lane,    e 0..7 -> K = e,     e 8..15 -> K = 16+(e-8)
//   lane 16-31: N = lane-16, e 0..7 -> K = 8+e,   e 8..15 -> K = 24+(e-8)
// packed[(nt*KT + kt)*512 + lane*16 + e] = W[kt*32 + (lane>>4)*8 + (e>>3)*16 + (e&7)][nt*16 + (lane&15)]
__device__ __forceinline__ void pack_one(const float* __restrict__ W,
                                         _Float16* __restrict__ dst,
                                         int K, int Nn, int tid, int nth) {
    const int KT = K / 32, NT = Nn / 16;
    const int total = NT * KT * 512;
    for (int idx = tid; idx < total; idx += nth) {
        int bi   = idx >> 9;
        int rem  = idx & 511;
        int lane = rem >> 4;
        int e    = rem & 15;
        int nt   = bi / KT, kt = bi % KT;
        int n    = nt * 16 + (lane & 15);
        int k    = kt * 32 + ((lane >> 4) << 3) + ((e >> 3) << 4) + (e & 7);
        dst[idx] = (_Float16)W[(size_t)k * Nn + n];
    }
}

__global__ void __launch_bounds__(256) pack_weights(
    const float* __restrict__ W1, const float* __restrict__ W2,
    const float* __restrict__ W3, const float* __restrict__ W4,
    _Float16* __restrict__ pw1, _Float16* __restrict__ pw2,
    _Float16* __restrict__ pw3, _Float16* __restrict__ pw4) {
    int tid = blockIdx.x * blockDim.x + threadIdx.x;
    int nth = gridDim.x * blockDim.x;
    pack_one(W1, pw1, DDIM, DDIM, tid, nth);
    pack_one(W2, pw2, DDIM, DDIM, tid, nth);
    pack_one(W3, pw3, DDIM, DDIM, tid, nth);
    pack_one(W4, pw4, DDIM, DOUT, tid, nth);
}

// --------------------------- WMMA helpers ----------------------------------
// A-operand from an LDS f16 row-major tile (ISA 7.12.2, 16-bit A 16x32):
//   lane 0-15 : M = lane,    e 0..7 -> K=k0+e,     e 8..15 -> K=k0+16+(e-8)
//   lane 16-31: M = lane-16, e 0..7 -> K=k0+8+e,   e 8..15 -> K=k0+24+(e-8)
__device__ __forceinline__ v16h load_a(const _Float16* buf, int rowBase,
                                       int mrow, int hi, int kt) {
    const _Float16* p = buf + (rowBase + mrow) * PITCH + kt * 32 + hi * 8;
    v8h lo = *(const v8h*)p;
    v8h hh = *(const v8h*)(p + 16);
    return __builtin_shufflevector(lo, hh, 0,1,2,3,4,5,6,7,8,9,10,11,12,13,14,15);
}

__device__ __forceinline__ v16h load_b(const _Float16* __restrict__ pw,
                                       int blockIdxB, int lane) {
    return *(const v16h*)(pw + (size_t)blockIdxB * 512 + lane * 16);
}

// Software-pipelined tile GEMM: A fully register-resident, B double-buffered
// one k-step ahead of the WMMA issue so loads retire behind matrix-pipe work.
template <int NT>
__device__ __forceinline__ void gemm_tile(const _Float16* buf, int rowBase,
                                          int mrow, int hi, int lane,
                                          const _Float16* __restrict__ pw,
                                          v8f acc[NT]) {
    v16h a[KTILES];
#pragma unroll
    for (int kt = 0; kt < KTILES; ++kt)
        a[kt] = load_a(buf, rowBase, mrow, hi, kt);

    v16h b[NT];
#pragma unroll
    for (int nt = 0; nt < NT; ++nt)
        b[nt] = load_b(pw, nt * KTILES, lane);

#pragma unroll
    for (int kt = 0; kt < KTILES; ++kt) {
        v16h bn[NT];
        if (kt + 1 < KTILES) {
#pragma unroll
            for (int nt = 0; nt < NT; ++nt)
                bn[nt] = load_b(pw, nt * KTILES + kt + 1, lane);
        }
#pragma unroll
        for (int nt = 0; nt < NT; ++nt)
            acc[nt] = __builtin_amdgcn_wmma_f32_16x16x32_f16(
                false, a[kt], false, b[nt], (short)0, acc[nt], false, false);
        if (kt + 1 < KTILES) {
#pragma unroll
            for (int nt = 0; nt < NT; ++nt)
                b[nt] = bn[nt];
        }
    }
}

// --------------- fused local MLP + sorted segment-sum ----------------------
__global__ void __launch_bounds__(THREADS) deepset_local(
    const float* __restrict__ x, const int* __restrict__ batch,
    const _Float16* __restrict__ pw1, const float* __restrict__ b1,
    const _Float16* __restrict__ pw2, const float* __restrict__ b2,
    float* __restrict__ s, int n) {
    __shared__ _Float16 buf[ROWS * PITCH];   // xtile -> h1 -> h2 (per-wave rows)
    __shared__ int sbatch[ROWS];

    const int t   = threadIdx.x;
    const int gr0 = blockIdx.x * ROWS;

    // stage x tile: f32 -> f16 in LDS (zero-pad rows past N)
    for (int i = t; i < ROWS * (DDIM / 4); i += THREADS) {
        int r    = i >> 5;            // DDIM/4 == 32 float4 per row
        int c4   = (i & 31) * 4;
        int grow = gr0 + r;
        v4h h4;
        if (grow < n) {
            const float4 v = ((const float4*)(x + (size_t)grow * DDIM))[c4 >> 2];
            h4 = (v4h){(_Float16)v.x, (_Float16)v.y, (_Float16)v.z, (_Float16)v.w};
        } else {
            h4 = (v4h){(_Float16)0.f, (_Float16)0.f, (_Float16)0.f, (_Float16)0.f};
        }
        *(v4h*)(buf + r * PITCH + c4) = h4;
    }
    if (t < ROWS) {
        int grow  = gr0 + t;
        sbatch[t] = (grow < n) ? batch[grow] : -1;
    }
    __syncthreads();

    const int lane = t & 31;
    const int wave = t >> 5;
    const int mrow = lane & 15;
    const int hi   = lane >> 4;
    const int R    = wave * 16;       // this wave's private 16-row band

    // GEMM1 + bias + relu -> h1 (overwrite own rows, DS in-order per wave)
    {
        v8f acc[8] = {};
        gemm_tile<8>(buf, R, mrow, hi, lane, pw1, acc);
#pragma unroll
        for (int nt = 0; nt < 8; ++nt) {
            float bias = b1[nt * 16 + mrow];
#pragma unroll
            for (int r = 0; r < 8; ++r) {
                float v = acc[nt][r] + bias;        // C/D: VGPR r -> M=r+8*hi, N=mrow
                v = v > 0.f ? v : 0.f;
                buf[(R + r + hi * 8) * PITCH + nt * 16 + mrow] = (_Float16)v;
            }
        }
    }
    // GEMM2 + bias -> h2 (f16 in same rows)
    {
        v8f acc[8] = {};
        gemm_tile<8>(buf, R, mrow, hi, lane, pw2, acc);
#pragma unroll
        for (int nt = 0; nt < 8; ++nt) {
            float bias = b2[nt * 16 + mrow];
#pragma unroll
            for (int r = 0; r < 8; ++r) {
                buf[(R + r + hi * 8) * PITCH + nt * 16 + mrow] =
                    (_Float16)(acc[nt][r] + bias);
            }
        }
    }
    __syncthreads();

    // run-length segment reduction over the sorted batch ids, then atomics
    const int col  = t & 127;
    const int half = t >> 7;
    float a  = 0.f;
    int  cur = -1;
    for (int r = half * 64; r < half * 64 + 64; ++r) {
        int grow = gr0 + r;
        if (grow >= n) break;
        int bidx = sbatch[r];
        if (bidx != cur) {
            if (cur >= 0) atomicAdd(&s[(size_t)cur * DDIM + col], a);
            cur = bidx; a = 0.f;
        }
        a += (float)buf[r * PITCH + col];
    }
    if (cur >= 0) atomicAdd(&s[(size_t)cur * DDIM + col], a);
}

// --------------------------- global MLP ------------------------------------
__global__ void __launch_bounds__(THREADS) deepset_global(
    const float* __restrict__ s,
    const _Float16* __restrict__ pw3, const float* __restrict__ b3,
    const _Float16* __restrict__ pw4, const float* __restrict__ b4,
    float* __restrict__ out, int nb) {
    __shared__ _Float16 buf[ROWS * PITCH];
    const int t   = threadIdx.x;
    const int gr0 = blockIdx.x * ROWS;

    for (int i = t; i < ROWS * (DDIM / 4); i += THREADS) {
        int r    = i >> 5;
        int c4   = (i & 31) * 4;
        int grow = gr0 + r;
        v4h h4;
        if (grow < nb) {
            const float4 v = ((const float4*)(s + (size_t)grow * DDIM))[c4 >> 2];
            h4 = (v4h){(_Float16)v.x, (_Float16)v.y, (_Float16)v.z, (_Float16)v.w};
        } else {
            h4 = (v4h){(_Float16)0.f, (_Float16)0.f, (_Float16)0.f, (_Float16)0.f};
        }
        *(v4h*)(buf + r * PITCH + c4) = h4;
    }
    __syncthreads();

    const int lane = t & 31;
    const int wave = t >> 5;
    const int mrow = lane & 15;
    const int hi   = lane >> 4;
    const int R    = wave * 16;

    {   // GEMM3 + bias + relu -> f16 in own rows
        v8f acc[8] = {};
        gemm_tile<8>(buf, R, mrow, hi, lane, pw3, acc);
#pragma unroll
        for (int nt = 0; nt < 8; ++nt) {
            float bias = b3[nt * 16 + mrow];
#pragma unroll
            for (int r = 0; r < 8; ++r) {
                float v = acc[nt][r] + bias;
                v = v > 0.f ? v : 0.f;
                buf[(R + r + hi * 8) * PITCH + nt * 16 + mrow] = (_Float16)v;
            }
        }
    }
    {   // GEMM4 + bias -> f32 out
        v8f acc[4] = {};
        gemm_tile<4>(buf, R, mrow, hi, lane, pw4, acc);
#pragma unroll
        for (int nt = 0; nt < 4; ++nt) {
            float bias = b4[nt * 16 + mrow];
#pragma unroll
            for (int r = 0; r < 8; ++r) {
                int grow = gr0 + R + r + hi * 8;
                if (grow < nb)
                    out[(size_t)grow * DOUT + nt * 16 + mrow] = acc[nt][r] + bias;
            }
        }
    }
}

// --------------------------- launcher --------------------------------------
extern "C" void kernel_launch(void* const* d_in, const int* in_sizes, int n_in,
                              void* d_out, int out_size, void* d_ws, size_t ws_size,
                              hipStream_t stream) {
    const float* x     = (const float*)d_in[0];
    const int*   batch = (const int*)d_in[1];
    const float* W1    = (const float*)d_in[2];
    const float* b1    = (const float*)d_in[3];
    const float* W2    = (const float*)d_in[4];
    const float* b2    = (const float*)d_in[5];
    const float* W3    = (const float*)d_in[6];
    const float* b3    = (const float*)d_in[7];
    const float* W4    = (const float*)d_in[8];
    const float* b4    = (const float*)d_in[9];

    const int n  = in_sizes[0] / DDIM;   // N rows
    const int nb = out_size / DOUT;      // B segments

    // workspace layout: packed f16 weights then f32 segment sums
    _Float16* pw1 = (_Float16*)d_ws;                 // 128x128 -> 16384 halves
    _Float16* pw2 = pw1 + 16384;
    _Float16* pw3 = pw2 + 16384;
    _Float16* pw4 = pw3 + 16384;                     // 128x64  ->  8192 halves
    float*    s   = (float*)(pw4 + 8192);            // [B,128] f32

    hipMemsetAsync(s, 0, (size_t)nb * DDIM * sizeof(float), stream);
    pack_weights<<<32, 256, 0, stream>>>(W1, W2, W3, W4, pw1, pw2, pw3, pw4);

    const int grid1 = (n + ROWS - 1) / ROWS;
    deepset_local<<<grid1, THREADS, 0, stream>>>(x, batch, pw1, b1, pw2, b2, s, n);

    const int grid2 = (nb + ROWS - 1) / ROWS;
    deepset_global<<<grid2, THREADS, 0, stream>>>(s, pw3, b3, pw4, b4,
                                                  (float*)d_out, nb);
}
